// GCNModel_32444182954792
// MI455X (gfx1250) — matrix operand order repaired and checked by reading, not compiled, not run on hardware
//
#include <hip/hip_runtime.h>
#include <hip/hip_bf16.h>

#define N_X 32640
#define N_S 16320
#define NE  65280
#define D_IN 768
#define GDIM 256
#define N_UPOS 17
#define POS_DIM 4
#define N_LABEL 5

typedef __attribute__((ext_vector_type(16))) __bf16 v16bf;
typedef __attribute__((ext_vector_type(2)))  __bf16 v2bf;
typedef __attribute__((ext_vector_type(8)))  float  f32x8;
typedef __attribute__((ext_vector_type(4)))  int    v4i;

union FragU {
    unsigned int u[8];
    v16bf v;
};

// pack two f32 -> packed bf16x2; native fptrunc lets the backend pick
// v_cvt_pk_bf16_f32-class ops (RNE either way)
__device__ inline unsigned int cvt2bf(float lo, float hi) {
#if __has_builtin(__builtin_amdgcn_cvt_pk_bf16_f32)
    union { v2bf v; unsigned int u; } c;
    c.v = __builtin_amdgcn_cvt_pk_bf16_f32(lo, hi);
    return c.u;
#else
    union { v2bf v; unsigned int u; } c;
    c.v.x = (__bf16)lo;
    c.v.y = (__bf16)hi;
    return c.u;
#endif
}

#if __has_builtin(__builtin_amdgcn_global_load_async_to_lds_b128)
#define HAVE_ASYNC_LDS 1
#else
#define HAVE_ASYNC_LDS 0
#endif

__device__ inline void wait_async_lds() {
#if __has_builtin(__builtin_amdgcn_s_wait_asynccnt)
    __builtin_amdgcn_s_wait_asynccnt(0);
#else
    asm volatile("s_wait_asynccnt 0x0" ::: "memory");
#endif
}

// ---------------------------------------------------------------------------
// degree / normalization kernels
// ---------------------------------------------------------------------------
__global__ void k_deg_init(float* deg) {
    int i = blockIdx.x * blockDim.x + threadIdx.x;
    if (i < N_S) deg[i] = 1.0f;                      // self loop
}

__global__ void k_deg_scatter(float* deg, const int* __restrict__ dsts) {
    int e = blockIdx.x * blockDim.x + threadIdx.x;
    if (e < NE) atomicAdd(&deg[dsts[e]], 1.0f);
}

__global__ void k_rsqrt(float* deg) {
    int i = blockIdx.x * blockDim.x + threadIdx.x;
    if (i < N_S) deg[i] = rsqrtf(deg[i]);            // in-place -> dinv
}

// one-shot f32 -> packed bf16 shadow (after GEMM1)
__global__ void k_cvt_bf16(const float* __restrict__ h, unsigned int* __restrict__ hbf)
{
    const size_t g = (size_t)blockIdx.x * blockDim.x + threadIdx.x;
    const float2 p = *(const float2*)(h + 2 * g);
    hbf[g] = cvt2bf(p.x, p.y);
}

// ---------------------------------------------------------------------------
// WMMA GEMM:  C[M x 256] = A[M x K] @ B[K x 256] (+bias)
// 256 threads (8 waves), tile 64(M) x 128(N), K-step 32, double-buffered LDS.
// POOL:  A = mean-pooled f32 x rows, converted to bf16 at staging.
// !POOL: A = packed-bf16 h shadow, raw async global->LDS copy (4KB/tile).
// B staged bf16 transposed [n][k]; all fragments are pure ds_load_b32/b128.
// ---------------------------------------------------------------------------
template <bool POOL>
__global__ void __launch_bounds__(256)
k_wmma_gemm(const float* __restrict__ Af, const unsigned short* __restrict__ Abf,
            const float* __restrict__ B, const float* __restrict__ bias,
            float* __restrict__ C, int K, int lda)
{
    __shared__ unsigned short lAb[2][64 * 32];       // [m][k] bf16
    __shared__ unsigned short lBu[2][128 * 32];      // [n][k] bf16 (transposed)

    const int tid  = threadIdx.x;
    const int wid  = tid >> 5;                       // 0..7
    const int lane = tid & 31;
    const int hl   = lane >> 4;
    const int l15  = lane & 15;
    const int wm   = wid & 3;                        // wave M slot (0..3)
    const int wn   = wid >> 2;                       // wave N half (0..1)

    const int mBase = blockIdx.x * 64;
    const int nBase = blockIdx.y * 128;

    f32x8 acc[4] = {};

    const int kTiles = K >> 5;

    // ---------------- tile staging ----------------
    auto stage = [&](int kt, int buf) {
        const int kBase = kt << 5;
        // ---- A tile: 64 rows x 32 k, bf16 ----
        if constexpr (POOL) {
            // fused subword mean-pool: avg rows 2m, 2m+1 of x, cvt to bf16
            const int r  = tid >> 2;                 // 0..63
            const int cb = (tid & 3) << 3;           // 0,8,16,24
            const float* p0 = Af + (size_t)(2 * (mBase + r)) * lda + kBase + cb;
            const float* p1 = p0 + lda;
            const float4 a0 = *(const float4*)(p0);
            const float4 a1 = *(const float4*)(p0 + 4);
            const float4 b0 = *(const float4*)(p1);
            const float4 b1 = *(const float4*)(p1 + 4);
            uint4 pk;
            pk.x = cvt2bf(0.5f * (a0.x + b0.x), 0.5f * (a0.y + b0.y));
            pk.y = cvt2bf(0.5f * (a0.z + b0.z), 0.5f * (a0.w + b0.w));
            pk.z = cvt2bf(0.5f * (a1.x + b1.x), 0.5f * (a1.y + b1.y));
            pk.w = cvt2bf(0.5f * (a1.z + b1.z), 0.5f * (a1.w + b1.w));
            *(uint4*)&lAb[buf][r * 32 + cb] = pk;
            if (kt + 1 < kTiles)                     // keep next rows warm
                __builtin_prefetch(p0 + 32, 0, 0);
        } else {
            // raw async copy of packed-bf16 A: 16B (8 bf16) per thread
            const int r  = tid >> 2;                 // 0..63
            const int cb = (tid & 3) << 3;           // bf16 col 0,8,16,24
            const unsigned short* src = Abf + (size_t)(mBase + r) * lda + kBase + cb;
            unsigned short* dst = &lAb[buf][r * 32 + cb];
#if HAVE_ASYNC_LDS
            __builtin_amdgcn_global_load_async_to_lds_b128(
                (v4i*)src, (v4i*)dst, 0, 0);
#else
            *(uint4*)dst = *(const uint4*)src;
#endif
        }
        // ---- B tile: k-pairs packed -> lBu[n*32 + kp] ----
        {
            const int kp = (tid >> 4) << 1;          // 0,2,..,30
            const int nb = (tid & 15) << 3;          // 0..120
            const float* s0 = B + (size_t)(kBase + kp) * GDIM + nBase + nb;
            const float* s1 = s0 + GDIM;             // row kp+1
            const float4 a0 = *(const float4*)(s0);
            const float4 a1 = *(const float4*)(s0 + 4);
            const float4 b0 = *(const float4*)(s1);
            const float4 b1 = *(const float4*)(s1 + 4);
            unsigned short* base = &lBu[buf][nb * 32 + kp];
            *(unsigned int*)(base + 0 * 32) = cvt2bf(a0.x, b0.x);
            *(unsigned int*)(base + 1 * 32) = cvt2bf(a0.y, b0.y);
            *(unsigned int*)(base + 2 * 32) = cvt2bf(a0.z, b0.z);
            *(unsigned int*)(base + 3 * 32) = cvt2bf(a0.w, b0.w);
            *(unsigned int*)(base + 4 * 32) = cvt2bf(a1.x, b1.x);
            *(unsigned int*)(base + 5 * 32) = cvt2bf(a1.y, b1.y);
            *(unsigned int*)(base + 6 * 32) = cvt2bf(a1.z, b1.z);
            *(unsigned int*)(base + 7 * 32) = cvt2bf(a1.w, b1.w);
        }
    };

    stage(0, 0);

    for (int kt = 0; kt < kTiles; ++kt) {
        const int buf = kt & 1;
#if HAVE_ASYNC_LDS
        if constexpr (!POOL) wait_async_lds();
#endif
        __syncthreads();
        if (kt + 1 < kTiles) stage(kt + 1, buf ^ 1);

        // ---- load A fragment + all 4 B fragments, then WMMA back-to-back --
        FragU a, bfr[4];
        const unsigned short* arow = &lAb[buf][(16 * wm + l15) * 32];
        #pragma unroll
        for (int i = 0; i < 8; ++i) {
            const int kp = ((i < 4) ? 2 * i : 2 * i + 8) + 8 * hl;
            a.u[i] = *(const unsigned int*)(arow + kp);
        }
        #pragma unroll
        for (int j = 0; j < 4; ++j) {
            const unsigned short* nrow = &lBu[buf][(64 * wn + 16 * j + l15) * 32];
            #pragma unroll
            for (int v = 0; v < 8; ++v)
                bfr[j].u[v] = *(const unsigned int*)(nrow + ((v < 4) ? 2 * v : 2 * v + 8) + 8 * hl);
        }
        #pragma unroll
        for (int j = 0; j < 4; ++j)
            acc[j] = __builtin_amdgcn_wmma_f32_16x16x32_bf16(
                false, a.v, false, bfr[j].v, (short)0, acc[j], false, false);
    }

    // ---- epilogue: f32 C/D layout -> m = r + 8*half, n = lane&15 ----
    #pragma unroll
    for (int j = 0; j < 4; ++j) {
        const int n = nBase + 64 * wn + 16 * j + l15;
        const float bv = bias ? bias[n] : 0.0f;
        #pragma unroll
        for (int r = 0; r < 8; ++r) {
            const int m = mBase + 16 * wm + r + 8 * hl;
            C[(size_t)m * GDIM + n] = acc[j][r] + bv;
        }
    }
}

// ---------------------------------------------------------------------------
// graph aggregation
// ---------------------------------------------------------------------------
__global__ void k_self_init(float* __restrict__ agg, const float* __restrict__ hw,
                            const float* __restrict__ dinv)
{
    const size_t idx = (size_t)blockIdx.x * blockDim.x + threadIdx.x;
    const int i = (int)(idx >> 8);
    const float di = dinv[i];
    agg[idx] = hw[idx] * di * di;                    // self-loop, plain store
}

__global__ void k_edge_scatter(float* __restrict__ agg, const float* __restrict__ hw,
                               const float* __restrict__ dinv,
                               const int* __restrict__ srcs, const int* __restrict__ dsts)
{
    const int e = blockIdx.x;
    const int s = srcs[e], d = dsts[e];
    const float c = dinv[s] * dinv[d];
    const int t = threadIdx.x;                       // 256 threads = 256 feats
    atomicAdd(&agg[(size_t)d * GDIM + t], hw[(size_t)s * GDIM + t] * c);
}

// h = relu(agg + b); also emit packed-bf16 shadow for the next GEMM's A
__global__ void k_finalize(float* __restrict__ h, unsigned int* __restrict__ hbf,
                           const float* __restrict__ agg, const float* __restrict__ b)
{
    const size_t g   = (size_t)blockIdx.x * blockDim.x + threadIdx.x;
    const size_t idx = 2 * g;
    const float v0 = fmaxf(agg[idx]     + b[idx & (GDIM - 1)],       0.0f);
    const float v1 = fmaxf(agg[idx + 1] + b[(idx + 1) & (GDIM - 1)], 0.0f);
    float2 p; p.x = v0; p.y = v1;
    *(float2*)&h[idx] = p;
    hbf[g] = cvt2bf(v0, v1);
}

// ---------------------------------------------------------------------------
// classifier: one wave per token, Wcls staged in LDS, fused log-softmax
// ---------------------------------------------------------------------------
#define CLS_K (D_IN + GDIM + POS_DIM)                // 1028

__global__ void __launch_bounds__(256)
k_classify(const float* __restrict__ x, const float* __restrict__ h,
           const int* __restrict__ upos_ids, const float* __restrict__ upt,
           const float* __restrict__ Wcls, const float* __restrict__ bcls,
           float* __restrict__ out)
{
    __shared__ float sW[CLS_K * N_LABEL];
    __shared__ float sB[N_LABEL];

    for (int i = threadIdx.x; i < CLS_K * N_LABEL; i += 256) sW[i] = Wcls[i];
    if (threadIdx.x < N_LABEL) sB[threadIdx.x] = bcls[threadIdx.x];
    __syncthreads();

    const int wid  = threadIdx.x >> 5;
    const int lane = threadIdx.x & 31;
    const int t    = blockIdx.x * 8 + wid;           // token row
    const int w    = t >> 1;                         // stanza word (seg = j//2)

    float acc[N_LABEL] = {0.f, 0.f, 0.f, 0.f, 0.f};

    for (int k = lane; k < D_IN; k += 32) {
        const float v = fmaxf(x[(size_t)t * D_IN + k], 0.0f);
        #pragma unroll
        for (int j = 0; j < N_LABEL; ++j) acc[j] += v * sW[k * N_LABEL + j];
    }
    for (int k = lane; k < GDIM; k += 32) {
        const float v = h[(size_t)w * GDIM + k];     // already relu'd
        #pragma unroll
        for (int j = 0; j < N_LABEL; ++j) acc[j] += v * sW[(D_IN + k) * N_LABEL + j];
    }
    const int uid = upos_ids[w];
    for (int k = lane; k < POS_DIM; k += 32) {
        const float v = fmaxf(upt[uid * POS_DIM + k], 0.0f);
        #pragma unroll
        for (int j = 0; j < N_LABEL; ++j) acc[j] += v * sW[(D_IN + GDIM + k) * N_LABEL + j];
    }
    #pragma unroll
    for (int j = 0; j < N_LABEL; ++j) {
        #pragma unroll
        for (int m = 16; m >= 1; m >>= 1) acc[j] += __shfl_xor(acc[j], m, 32);
    }

    if (lane == 0) {
        float lg[N_LABEL], mx = -1e30f;
        #pragma unroll
        for (int j = 0; j < N_LABEL; ++j) { lg[j] = acc[j] + sB[j]; mx = fmaxf(mx, lg[j]); }
        float s = 0.f;
        #pragma unroll
        for (int j = 0; j < N_LABEL; ++j) s += __expf(lg[j] - mx);
        const float lse = __logf(s);
        #pragma unroll
        for (int j = 0; j < N_LABEL; ++j) out[(size_t)t * N_LABEL + j] = lg[j] - mx - lse;
    }
}

// ---------------------------------------------------------------------------
extern "C" void kernel_launch(void* const* d_in, const int* in_sizes, int n_in,
                              void* d_out, int out_size, void* d_ws, size_t ws_size,
                              hipStream_t stream)
{
    const float* x        = (const float*)d_in[0];
    // d_in[1] = seg (structurally j//2, folded into indexing)
    const int*   edge     = (const int*)d_in[2];
    const int*   upos_ids = (const int*)d_in[3];
    const float* Wc       = (const float*)d_in[4];
    const float* bc       = (const float*)d_in[5];
    const float* gcn_W    = (const float*)d_in[6];
    const float* gcn_b    = (const float*)d_in[7];
    const float* upt      = (const float*)d_in[8];
    const float* Wcls     = (const float*)d_in[9];
    const float* bcls     = (const float*)d_in[10];
    float*       out      = (float*)d_out;

    const int* srcs = edge;
    const int* dsts = edge + NE;

    float*        h    = (float*)d_ws;               // [N_S, 256] f32
    float*        hw   = h   + (size_t)N_S * GDIM;   // [N_S, 256] f32
    float*        agg  = hw  + (size_t)N_S * GDIM;   // [N_S, 256] f32
    float*        dinv = agg + (size_t)N_S * GDIM;   // [N_S]      f32
    unsigned int* hbf  = (unsigned int*)(dinv + N_S);// [N_S, 128] bf16x2

    // symmetric-normalization coefficients (shared by both layers)
    k_deg_init   <<<(N_S + 255) / 256, 256, 0, stream>>>(dinv);
    k_deg_scatter<<<NE / 256,          256, 0, stream>>>(dinv, dsts);
    k_rsqrt      <<<(N_S + 255) / 256, 256, 0, stream>>>(dinv);

    // h = meanpool(x) @ Wc + bc   (pooling fused into A-tile load)
    dim3 gg(N_S / 64, GDIM / 128);
    k_wmma_gemm<true><<<gg, 256, 0, stream>>>(x, nullptr, Wc, bc, h, D_IN, D_IN);
    k_cvt_bf16<<<(N_S * GDIM / 2) / 256, 256, 0, stream>>>(h, hbf);

    for (int l = 0; l < 2; ++l) {
        k_wmma_gemm<false><<<gg, 256, 0, stream>>>(nullptr, (const unsigned short*)hbf,
                                                   gcn_W + (size_t)l * GDIM * GDIM,
                                                   nullptr, hw, GDIM, GDIM);
        k_self_init   <<<(N_S * GDIM) / 256,     256, 0, stream>>>(agg, hw, dinv);
        k_edge_scatter<<<NE,                     256, 0, stream>>>(agg, hw, dinv, srcs, dsts);
        k_finalize    <<<(N_S * GDIM / 2) / 256, 256, 0, stream>>>(h, hbf, agg, gcn_b + l * GDIM);
    }

    k_classify<<<N_X / 8, 256, 0, stream>>>(x, h, upos_ids, upt, Wcls, bcls, out);
}
